// CausalSelfAttention_75634374083254
// MI455X (gfx1250) — matrix operand order repaired and checked by reading, not compiled, not run on hardware
//
#include <hip/hip_runtime.h>

// ---------------------------------------------------------------------------
// Problem constants (match reference)
// ---------------------------------------------------------------------------
#define DMODEL  1024
#define NHEAD   16
#define HDIM    64
#define CTXLEN  2048
#define BATCH   2
#define MTOT    (BATCH * CTXLEN)        // 4096 token rows

typedef __attribute__((ext_vector_type(16))) __bf16 bfrag;   // 8 VGPRs / lane
typedef __attribute__((ext_vector_type(8)))  float  v8f;     // 8 VGPRs / lane

struct B32 { uint4 lo, hi; };           // 32 bytes == one bf16 fragment

// round-to-nearest-even fp32 -> bf16 (as raw u16)
__device__ __forceinline__ unsigned short f2bf(float f) {
    unsigned u = __builtin_bit_cast(unsigned, f);
    unsigned r = u + 0x7FFFu + ((u >> 16) & 1u);
    return (unsigned short)(r >> 16);
}

// A-matrix 16x32 bf16 fragment (k order used here, same as B):
//   lanes 0-15 : M=lane,    K=k0..k0+15   (contiguous)
//   lanes16-31 : M=lane-16, K=k0+16..+31  (contiguous)
__device__ __forceinline__ bfrag load_a_frag(const unsigned short* __restrict__ src,
                                             int ld, int m0, int k0, int l, int h) {
    const char* base = (const char*)(src + (long)(m0 + l) * ld + k0 + h * 16);
    B32 r;
    r.lo = *(const uint4*)(base);
    r.hi = *(const uint4*)(base + 16);   // +8 bf16 elements
    return __builtin_bit_cast(bfrag, r);
}

// B-matrix 32x16 bf16 fragment, B(k,n) = src[(n0+n)*ld + k0+k]:
//   lanes 0-15 : N=lane,    K=k0..k0+15   (contiguous)
//   lanes16-31 : N=lane-16, K=k0+16..+31  (contiguous)
__device__ __forceinline__ bfrag load_b_frag(const unsigned short* __restrict__ src,
                                             int ld, int n0, int k0, int l, int h) {
    const char* base = (const char*)(src + (long)(n0 + l) * ld + k0 + h * 16);
    B32 r;
    r.lo = *(const uint4*)(base);
    r.hi = *(const uint4*)(base + 16);
    return __builtin_bit_cast(bfrag, r);
}

#define WMMA_BF16(A, B, C) \
    __builtin_amdgcn_wmma_f32_16x16x32_bf16(false, (A), false, (B), (short)0, (C), false, false)

// ---------------------------------------------------------------------------
// 1) fp32 -> bf16 conversion, 4 elements / thread (all sizes %4 == 0)
// ---------------------------------------------------------------------------
__global__ void k_conv_bf16(const float* __restrict__ in,
                            unsigned short* __restrict__ out, int n4) {
    int i = blockIdx.x * blockDim.x + threadIdx.x;
    if (i < n4) {
        float4 v = ((const float4*)in)[i];
        ushort4 o;
        o.x = f2bf(v.x); o.y = f2bf(v.y); o.z = f2bf(v.z); o.w = f2bf(v.w);
        ((ushort4*)out)[i] = o;
    }
}

// ---------------------------------------------------------------------------
// GEMM core: 32x64 tile per wave, K-loop unrolled x2 with ping-pong buffers
// (buffers are re-assigned by loads, never copied -> no mov chains, and each
//  WMMA batch waits on loads issued one batch earlier).
// ---------------------------------------------------------------------------
#define GEMM_PINGPONG_BODY(Asrc, Bsrc, LD)                                        \
    v8f acc[2][4] = {};                                                           \
    bfrag a0A = load_a_frag(Asrc, LD, m0,      0, l, h);                          \
    bfrag a1A = load_a_frag(Asrc, LD, m0 + 16, 0, l, h);                          \
    bfrag bA0 = load_b_frag(Bsrc, LD, n0,      0, l, h);                          \
    bfrag bA1 = load_b_frag(Bsrc, LD, n0 + 16, 0, l, h);                          \
    bfrag bA2 = load_b_frag(Bsrc, LD, n0 + 32, 0, l, h);                          \
    bfrag bA3 = load_b_frag(Bsrc, LD, n0 + 48, 0, l, h);                          \
    for (int k0 = 0; k0 < DMODEL; k0 += 64) {                                     \
        /* issue loads for k0+32 (always < DMODEL since DMODEL % 64 == 0) */      \
        bfrag a0B = load_a_frag(Asrc, LD, m0,      k0 + 32, l, h);                \
        bfrag a1B = load_a_frag(Asrc, LD, m0 + 16, k0 + 32, l, h);                \
        bfrag bB0 = load_b_frag(Bsrc, LD, n0,      k0 + 32, l, h);                \
        bfrag bB1 = load_b_frag(Bsrc, LD, n0 + 16, k0 + 32, l, h);                \
        bfrag bB2 = load_b_frag(Bsrc, LD, n0 + 32, k0 + 32, l, h);                \
        bfrag bB3 = load_b_frag(Bsrc, LD, n0 + 48, k0 + 32, l, h);                \
        /* compute with buffer A (k0) */                                          \
        acc[0][0] = WMMA_BF16(a0A, bA0, acc[0][0]);                               \
        acc[0][1] = WMMA_BF16(a0A, bA1, acc[0][1]);                               \
        acc[0][2] = WMMA_BF16(a0A, bA2, acc[0][2]);                               \
        acc[0][3] = WMMA_BF16(a0A, bA3, acc[0][3]);                               \
        acc[1][0] = WMMA_BF16(a1A, bA0, acc[1][0]);                               \
        acc[1][1] = WMMA_BF16(a1A, bA1, acc[1][1]);                               \
        acc[1][2] = WMMA_BF16(a1A, bA2, acc[1][2]);                               \
        acc[1][3] = WMMA_BF16(a1A, bA3, acc[1][3]);                               \
        /* issue loads for k0+64 into buffer A (guarded dummy on last iter) */    \
        const int kn = (k0 + 64 < DMODEL) ? k0 + 64 : 0;                          \
        a0A = load_a_frag(Asrc, LD, m0,      kn, l, h);                           \
        a1A = load_a_frag(Asrc, LD, m0 + 16, kn, l, h);                           \
        bA0 = load_b_frag(Bsrc, LD, n0,      kn, l, h);                           \
        bA1 = load_b_frag(Bsrc, LD, n0 + 16, kn, l, h);                           \
        bA2 = load_b_frag(Bsrc, LD, n0 + 32, kn, l, h);                           \
        bA3 = load_b_frag(Bsrc, LD, n0 + 48, kn, l, h);                           \
        /* compute with buffer B (k0+32) */                                       \
        acc[0][0] = WMMA_BF16(a0B, bB0, acc[0][0]);                               \
        acc[0][1] = WMMA_BF16(a0B, bB1, acc[0][1]);                               \
        acc[0][2] = WMMA_BF16(a0B, bB2, acc[0][2]);                               \
        acc[0][3] = WMMA_BF16(a0B, bB3, acc[0][3]);                               \
        acc[1][0] = WMMA_BF16(a1B, bB0, acc[1][0]);                               \
        acc[1][1] = WMMA_BF16(a1B, bB1, acc[1][1]);                               \
        acc[1][2] = WMMA_BF16(a1B, bB2, acc[1][2]);                               \
        acc[1][3] = WMMA_BF16(a1B, bB3, acc[1][3]);                               \
    }

// ---------------------------------------------------------------------------
// 2) QKV GEMM: qkv = x @ w_qkv^T, scattered into Q[B,H,T,D], K[B,H,T,D],
//    V^T[B,H,D,T] (bf16). M = 4096, N = 3072, K = 1024.
// ---------------------------------------------------------------------------
__global__ __launch_bounds__(256) void k_gemm_qkv(
        const unsigned short* __restrict__ xb,
        const unsigned short* __restrict__ wb,
        unsigned short* __restrict__ qws,
        unsigned short* __restrict__ kws,
        unsigned short* __restrict__ vtws) {
    const int lane = threadIdx.x & 31;
    const int l = lane & 15, h = lane >> 4;
    int gw = blockIdx.x * 8 + (threadIdx.x >> 5);   // 6144 waves
    int nt = gw % 48, mt = gw / 48;                 // mt 0..127
    int m0 = mt * 32, n0 = nt * 64;

    GEMM_PINGPONG_BODY(xb, wb, DMODEL)

    // D layout: VGPR r -> row h*8+r, col = lane&15
#pragma unroll
    for (int i = 0; i < 2; ++i) {
#pragma unroll
        for (int f = 0; f < 4; ++f) {
#pragma unroll
            for (int r = 0; r < 8; ++r) {
                int m = m0 + i * 16 + h * 8 + r;         // token row
                int o = n0 + f * 16 + l;                 // 0..3071
                int mat = o >> 10, c = o & (DMODEL - 1);
                int head = c >> 6, d = c & (HDIM - 1);
                int b_ = m >> 11, t = m & (CTXLEN - 1);
                int bh = b_ * NHEAD + head;
                unsigned short v = f2bf(acc[i][f][r]);
                if (mat == 0)      qws[((long)bh * CTXLEN + t) * HDIM + d] = v;
                else if (mat == 1) kws[((long)bh * CTXLEN + t) * HDIM + d] = v;
                else               vtws[((long)bh * HDIM + d) * CTXLEN + t] = v;
            }
        }
    }
}

// ---------------------------------------------------------------------------
// 3) Flash attention. One wave handles one (b,h, 16-query tile), streaming
//    keys 32 at a time with online softmax. scale = *sqrt(HDIM)* = 8 (the
//    reference MULTIPLIES by sqrt(hdim)). Software-pipelined: fragments for
//    iteration i+1 are issued at the top of iteration i; the rotation at the
//    bottom absorbs the load waits after a full iteration of compute.
// ---------------------------------------------------------------------------
__global__ __launch_bounds__(256) void k_attn(
        const unsigned short* __restrict__ qws,
        const unsigned short* __restrict__ kws,
        const unsigned short* __restrict__ vtws,
        unsigned short* __restrict__ yws) {
    __shared__ unsigned short pbuf[8][16 * 32];      // per-wave P tile (bf16)
    const int lane = threadIdx.x & 31;
    const int wid = threadIdx.x >> 5;
    const int l = lane & 15, h = lane >> 4;
    int gw = blockIdx.x * 8 + wid;                   // 4096 waves
    int qt = gw & 127;                               // q tile within (b,h)
    int bh = gw >> 7;                                // 0..31
    int t0 = qt * 16;

    const unsigned short* Q  = qws  + (long)bh * CTXLEN * HDIM;
    const unsigned short* K  = kws  + (long)bh * CTXLEN * HDIM;
    const unsigned short* Vt = vtws + (long)bh * HDIM * CTXLEN;

    bfrag qa0 = load_a_frag(Q, HDIM, t0, 0,  l, h);
    bfrag qa1 = load_a_frag(Q, HDIM, t0, 32, l, h);

    // prologue: fragments for kb = 0
    bfrag cb00 = load_b_frag(K, HDIM, 0,  0,  l, h);
    bfrag cb01 = load_b_frag(K, HDIM, 0,  32, l, h);
    bfrag cb10 = load_b_frag(K, HDIM, 16, 0,  l, h);
    bfrag cb11 = load_b_frag(K, HDIM, 16, 32, l, h);
    bfrag cvb0 = load_b_frag(Vt, CTXLEN, 0,  0, l, h);
    bfrag cvb1 = load_b_frag(Vt, CTXLEN, 16, 0, l, h);
    bfrag cvb2 = load_b_frag(Vt, CTXLEN, 32, 0, l, h);
    bfrag cvb3 = load_b_frag(Vt, CTXLEN, 48, 0, l, h);

    v8f o[4] = {};
    float rmax[8], rsum[8];
#pragma unroll
    for (int r = 0; r < 8; ++r) { rmax[r] = -__builtin_inff(); rsum[r] = 0.f; }

    const int kend = t0 + 16;                        // causal key bound (excl)
    for (int kb = 0; kb < kend; kb += 32) {
        // issue next iteration's loads first (dummy reload of kb=0 on last)
        const int kn = (kb + 32 < kend) ? kb + 32 : 0;
        bfrag nb00 = load_b_frag(K, HDIM, kn,      0,  l, h);
        bfrag nb01 = load_b_frag(K, HDIM, kn,      32, l, h);
        bfrag nb10 = load_b_frag(K, HDIM, kn + 16, 0,  l, h);
        bfrag nb11 = load_b_frag(K, HDIM, kn + 16, 32, l, h);
        bfrag nvb0 = load_b_frag(Vt, CTXLEN, 0,  kn, l, h);
        bfrag nvb1 = load_b_frag(Vt, CTXLEN, 16, kn, l, h);
        bfrag nvb2 = load_b_frag(Vt, CTXLEN, 32, kn, l, h);
        bfrag nvb3 = load_b_frag(Vt, CTXLEN, 48, kn, l, h);
        __builtin_prefetch(K  + (long)(kb + 64 + l + h * 16) * HDIM, 0, 0);
        __builtin_prefetch(Vt + (long)(l + h * 16) * CTXLEN + kb + 64, 0, 0);

        // S = Q K^T : two 16x16 n-tiles, K-dim = 64 -> 2 chained WMMAs each
        v8f s0 = {}, s1 = {};
        s0 = WMMA_BF16(qa0, cb00, s0);
        s0 = WMMA_BF16(qa1, cb01, s0);
        s1 = WMMA_BF16(qa0, cb10, s1);
        s1 = WMMA_BF16(qa1, cb11, s1);

        // scale + causal mask
        float sv0[8], sv1[8];
        const int col0 = kb + l, col1 = kb + 16 + l;
#pragma unroll
        for (int r = 0; r < 8; ++r) {
            int row = t0 + h * 8 + r;
            sv0[r] = (col0 <= row) ? s0[r] * 8.0f : -__builtin_inff();
            sv1[r] = (col1 <= row) ? s1[r] * 8.0f : -__builtin_inff();
        }
        // online softmax: row stats live across the 16-lane half groups
        float corr[8];
#pragma unroll
        for (int r = 0; r < 8; ++r) {
            float m = fmaxf(sv0[r], sv1[r]);
            m = fmaxf(m, __shfl_xor(m, 1));
            m = fmaxf(m, __shfl_xor(m, 2));
            m = fmaxf(m, __shfl_xor(m, 4));
            m = fmaxf(m, __shfl_xor(m, 8));
            float mnew = fmaxf(rmax[r], m);
            corr[r] = __expf(rmax[r] - mnew);
            rmax[r] = mnew;
            float p0 = __expf(sv0[r] - mnew);
            float p1 = __expf(sv1[r] - mnew);
            pbuf[wid][(h * 8 + r) * 32 + l]      = f2bf(p0);
            pbuf[wid][(h * 8 + r) * 32 + 16 + l] = f2bf(p1);
            float ps = p0 + p1;
            ps += __shfl_xor(ps, 1);
            ps += __shfl_xor(ps, 2);
            ps += __shfl_xor(ps, 4);
            ps += __shfl_xor(ps, 8);
            rsum[r] = rsum[r] * corr[r] + ps;
        }
#pragma unroll
        for (int dt = 0; dt < 4; ++dt)
#pragma unroll
            for (int r = 0; r < 8; ++r)
                o[dt][r] *= corr[r];

        // P (C/D layout) -> LDS -> A-fragment layout; same-wave, DS in order
        asm volatile("s_wait_dscnt 0" ::: "memory");
        bfrag pa = load_a_frag(&pbuf[wid][0], 32, 0, 0, l, h);
        o[0] = WMMA_BF16(pa, cvb0, o[0]);
        o[1] = WMMA_BF16(pa, cvb1, o[1]);
        o[2] = WMMA_BF16(pa, cvb2, o[2]);
        o[3] = WMMA_BF16(pa, cvb3, o[3]);

        // rotate: waits on the next-buffers land here, after a full iteration
        cb00 = nb00; cb01 = nb01; cb10 = nb10; cb11 = nb11;
        cvb0 = nvb0; cvb1 = nvb1; cvb2 = nvb2; cvb3 = nvb3;
    }

    // normalize and write y[token, c] bf16 (c = head*64 + d)
    int head = bh & (NHEAD - 1);
    int b_ = bh >> 4;
#pragma unroll
    for (int dt = 0; dt < 4; ++dt) {
#pragma unroll
        for (int r = 0; r < 8; ++r) {
            float val = o[dt][r] / rsum[r];
            long row = (long)b_ * CTXLEN + t0 + h * 8 + r;
            int  cc  = head * HDIM + dt * 16 + l;
            yws[row * DMODEL + cc] = f2bf(val);
        }
    }
}

// ---------------------------------------------------------------------------
// 4) Output projection: out = y @ w_proj^T + b_proj (fp32 out)
//    M = 4096, N = 1024, K = 1024.
// ---------------------------------------------------------------------------
__global__ __launch_bounds__(256) void k_gemm_proj(
        const unsigned short* __restrict__ yb,
        const unsigned short* __restrict__ wb,
        const float* __restrict__ bias,
        float* __restrict__ out) {
    const int lane = threadIdx.x & 31;
    const int l = lane & 15, h = lane >> 4;
    int gw = blockIdx.x * 8 + (threadIdx.x >> 5);    // 2048 waves
    int nt = gw & 15, mt = gw >> 4;                  // mt 0..127
    int m0 = mt * 32, n0 = nt * 64;

    GEMM_PINGPONG_BODY(yb, wb, DMODEL)

#pragma unroll
    for (int i = 0; i < 2; ++i) {
#pragma unroll
        for (int f = 0; f < 4; ++f) {
            float bv = bias[n0 + f * 16 + l];
#pragma unroll
            for (int r = 0; r < 8; ++r) {
                long row = m0 + i * 16 + h * 8 + r;
                out[row * DMODEL + n0 + f * 16 + l] = acc[i][f][r] + bv;
            }
        }
    }
}

// ---------------------------------------------------------------------------
// Host launcher
// ---------------------------------------------------------------------------
extern "C" void kernel_launch(void* const* d_in, const int* in_sizes, int n_in,
                              void* d_out, int out_size, void* d_ws, size_t ws_size,
                              hipStream_t stream) {
    const float* x      = (const float*)d_in[0];
    const float* w_qkv  = (const float*)d_in[1];
    const float* w_proj = (const float*)d_in[2];
    const float* b_proj = (const float*)d_in[3];
    float* out = (float*)d_out;

    // workspace layout (bytes)
    const size_t XB    = (size_t)MTOT * DMODEL * 2;          // 8 MB
    const size_t WQKVB = (size_t)3 * DMODEL * DMODEL * 2;    // 6 MB
    const size_t WPROJ = (size_t)DMODEL * DMODEL * 2;        // 2 MB
    const size_t QKV1  = (size_t)BATCH * NHEAD * CTXLEN * HDIM * 2; // 8 MB each
    char* ws = (char*)d_ws;
    unsigned short* xb     = (unsigned short*)(ws);
    unsigned short* wqkvb  = (unsigned short*)(ws + XB);
    unsigned short* wprojb = (unsigned short*)(ws + XB + WQKVB);
    unsigned short* qws    = (unsigned short*)(ws + XB + WQKVB + WPROJ);
    unsigned short* kws    = (unsigned short*)(ws + XB + WQKVB + WPROJ + QKV1);
    unsigned short* vtws   = (unsigned short*)(ws + XB + WQKVB + WPROJ + 2 * QKV1);
    // y aliases xb: x is fully consumed by the QKV GEMM before attention runs
    unsigned short* yws    = xb;

    // 1) conversions (vectorized x4)
    {
        int n4;
        n4 = (MTOT * DMODEL) / 4;
        k_conv_bf16<<<(n4 + 255) / 256, 256, 0, stream>>>(x, xb, n4);
        n4 = (3 * DMODEL * DMODEL) / 4;
        k_conv_bf16<<<(n4 + 255) / 256, 256, 0, stream>>>(w_qkv, wqkvb, n4);
        n4 = (DMODEL * DMODEL) / 4;
        k_conv_bf16<<<(n4 + 255) / 256, 256, 0, stream>>>(w_proj, wprojb, n4);
    }
    // 2) QKV GEMM: 128 m32-tiles * 48 n64-tiles = 6144 waves -> 768 blocks
    k_gemm_qkv<<<768, 256, 0, stream>>>(xb, wqkvb, qws, kws, vtws);
    // 3) attention: 32 (b,h) * 128 q-tiles = 4096 waves -> 512 blocks
    k_attn<<<512, 256, 0, stream>>>(qws, kws, vtws, yws);
    // 4) projection: 128 m32-tiles * 16 n64-tiles = 2048 waves -> 256 blocks
    k_gemm_proj<<<256, 256, 0, stream>>>(yws, wprojb, b_proj, out);
}
